// lstm_23192823398710
// MI455X (gfx1250) — matrix operand order repaired and checked by reading, not compiled, not run on hardware
//
#include <hip/hip_runtime.h>
#include <math.h>

// ---------------- types ----------------
typedef __attribute__((ext_vector_type(16))) __bf16 v16bf;
typedef __attribute__((ext_vector_type(8)))  __bf16 v8bf;
typedef __attribute__((ext_vector_type(8)))  float  v8f;

#define TT   256
#define BB   128
#define EE   512
#define HH   1024
#define KK   (EE + HH)      // 1536
#define G4H  (4 * HH)       // 4096
#define LBL  5
#define NCH  (KK / 32)      // 48 k-chunks

// f32 -> bf16 (round to nearest even), raw bits
static __device__ __forceinline__ unsigned short f2bf(float x) {
    union { float f; unsigned int u; } v; v.f = x;
    unsigned int u = v.u;
    u += 0x7fffu + ((u >> 16) & 1u);
    return (unsigned short)(u >> 16);
}

static __device__ __forceinline__ float sigm(float x) {
    return 1.0f / (1.0f + __expf(-x));
}

static __device__ __forceinline__ v16bf cat8(v8bf lo, v8bf hi) {
    v16bf r;
#pragma unroll
    for (int i = 0; i < 8; ++i) { r[i] = lo[i]; r[i + 8] = hi[i]; }
    return r;
}

// ---------------- 1) embedding gather -> bf16 (T,B,E) ----------------
__global__ __launch_bounds__(256) void gather_kernel(
    const int* __restrict__ inputs, const float* __restrict__ emb_table,
    unsigned short* __restrict__ embB)
{
    size_t id = (size_t)blockIdx.x * 256 + threadIdx.x;       // (t*B + b)*E + e
    if (id >= (size_t)TT * BB * EE) return;
    int e = (int)(id & (EE - 1));
    int b = (int)((id >> 9) & (BB - 1));
    int t = (int)(id >> 16);
    int tok = inputs[b * TT + t];                             // inputs is (B,T)
    embB[id] = f2bf(emb_table[(size_t)tok * EE + e]);
}

// ------- 2) pack WcatT[dir][n][k] = bf16( k<E ? Wx[k][n] : Wh[k-E][n] ) -------
__global__ __launch_bounds__(256) void pack_kernel(
    const float* __restrict__ Wxf, const float* __restrict__ Whf,
    const float* __restrict__ Wxb, const float* __restrict__ Whb,
    unsigned short* __restrict__ WT)
{
    size_t id = (size_t)blockIdx.x * 256 + threadIdx.x;
    if (id >= (size_t)2 * G4H * KK) return;
    int dir = (int)(id / ((size_t)G4H * KK));
    size_t rem = id - (size_t)dir * G4H * KK;
    int n = (int)(rem / KK);
    int k = (int)(rem % KK);
    const float* Wx = dir ? Wxb : Wxf;
    const float* Wh = dir ? Whb : Whf;
    float v = (k < EE) ? Wx[(size_t)k * G4H + n]
                       : Wh[(size_t)(k - EE) * G4H + n];
    WT[id] = f2bf(v);
}

// ---------------- 3) state init ----------------
__global__ __launch_bounds__(256) void init_kernel(
    float* __restrict__ Cst, unsigned short* __restrict__ Hb16,
    float* __restrict__ maxbuf)
{
    int id = blockIdx.x * 256 + threadIdx.x;
    if (id < 2 * BB * HH) { Cst[id] = 0.0f; maxbuf[id] = -INFINITY; }
    if (id < 2 * 2 * BB * HH) Hb16[id] = 0;   // both parity buffers
}

// -------- 4) one recurrent step, both directions, fully fused --------
// Block = 8 waves sharing (dir, hTile), spanning rowTile 0..7 so the four
// gate B-fragments (4 KB / 32-k chunk) are staged once per block in LDS,
// double-buffered (one barrier per chunk). Each wave owns a 16(batch)x16(h)
// tile, accumulates 4 gate tiles over K = E + H, then does the LSTM
// pointwise in-register.
__global__ __launch_bounds__(256) void step_kernel(
    const unsigned short* __restrict__ embB,   // (T,B,E) bf16
    const unsigned short* __restrict__ WT,     // (2,4096,1536) bf16
    const float* __restrict__ bhf, const float* __restrict__ bhb,
    float* __restrict__ Cst,                   // (2,B,H) f32
    unsigned short* __restrict__ Hb16,         // (2 parity, 2, B, H) bf16
    float* __restrict__ maxbuf,                // (B, 2H) f32 running max
    int t)
{
    __shared__ unsigned short sB[2][2048];     // 2 x 4 KB chunk buffers

    const int tid  = threadIdx.x;
    const int lane = tid & 31;
    const int rowTile = tid >> 5;              // wave id = rowTile 0..7
    const int dir   = blockIdx.x >> 6;         // 0..1
    const int hTile = blockIdx.x & 63;         // 0..63
    const int te    = dir ? (TT - 1 - t) : t;

    const bool hiHalf = (lane & 16) != 0;
    const int  m      = lane & 15;
    const int  rowA   = rowTile * 16 + m;      // batch row this lane supplies
    const int  nloc   = lane & 15;             // column within tile
    const int  koff   = hiHalf ? 8 : 0;

    const unsigned short* Wbase  = WT + (size_t)dir * G4H * KK;
    const unsigned short* embRow = embB + ((size_t)te * BB + rowA) * EE;
    const unsigned short* Hread  = Hb16 + (size_t)(t & 1) * (2 * BB * HH)
                                        + ((size_t)dir * BB + rowA) * HH;
    unsigned short* Hwrite = Hb16 + (size_t)((t + 1) & 1) * (2 * BB * HH);

    // cooperative weight staging: thread -> 8 halves (16 B) of the 4 KB chunk
    const int flat = tid * 8;                  // 0..2040
    const int gG   = flat >> 9;                // gate 0..3  (512 halves/gate)
    const int nG   = (flat >> 5) & 15;         // column in tile
    const int k0   = flat & 31;                // 0,8,16,24
    const unsigned short* wsrc =
        Wbase + (size_t)(gG * HH + hTile * 16 + nG) * KK + k0;

    v8f acc[4];
#pragma unroll
    for (int g = 0; g < 4; ++g)
#pragma unroll
        for (int i = 0; i < 8; ++i) acc[g][i] = 0.0f;

    // preload chunk 0 into buffer 0
    *(v8bf*)(&sB[0][flat]) = *(const v8bf*)(wsrc);

    v8bf pf;
#pragma unroll
    for (int i = 0; i < 8; ++i) pf[i] = (__bf16)0.0f;

    for (int i = 0; i < NCH; ++i) {
        __syncthreads();                       // sB[i&1] ready; prev reads done
        if (i + 1 < NCH)                       // prefetch next chunk (global)
            pf = *(const v8bf*)(wsrc + (i + 1) * 32);

        // ---- A fragment (16x32 bf16): lane half -> K {0-7,16-23}/{8-15,24-31}
        const int kc = i * 32;
        const int kb = kc + (hiHalf ? 8 : 0);
        const unsigned short* asrc =
            (kc < EE) ? (embRow + kb) : (Hread + (kb - EE));
        v16bf afrag = cat8(*(const v8bf*)(asrc),
                           *(const v8bf*)(asrc + 16));

        // ---- B fragments from LDS: lanes 0-15 -> K {0-7,16-23}, lanes 16-31 -> K {8-15,24-31}
        const unsigned short* sb = &sB[i & 1][0];
#pragma unroll
        for (int g = 0; g < 4; ++g) {
            const unsigned short* bsrc = sb + ((g * 16 + nloc) * 32 + koff);
            v16bf bfrag = cat8(*(const v8bf*)(bsrc),
                               *(const v8bf*)(bsrc + 16));
            acc[g] = __builtin_amdgcn_wmma_f32_16x16x32_bf16(
                false, afrag, false, bfrag, (short)0, acc[g], false, false);
        }

        if (i + 1 < NCH)                       // fill the other buffer
            *(v8bf*)(&sB[(i + 1) & 1][flat]) = pf;
    }

    // ---- fused LSTM pointwise: C/D layout lane L: N = L%16, VGPR r: M = r + 8*(L>=16)
    const float* bias = dir ? bhb : bhf;
    const int h    = hTile * 16 + nloc;
    const int mofs = hiHalf ? 8 : 0;
    const float bi = bias[h];
    const float bf = bias[HH + h];
    const float bo = bias[2 * HH + h];
    const float bc = bias[3 * HH + h];
#pragma unroll
    for (int r = 0; r < 8; ++r) {
        const int b = rowTile * 16 + mofs + r;
        float i_ = sigm(acc[0][r] + bi);
        float f_ = sigm(acc[1][r] + bf);
        float o_ = sigm(acc[2][r] + bo);
        if (dir) { i_ = sigm(i_); f_ = sigm(f_); o_ = sigm(o_); }  // double-sigmoid bwd
        const float c_ = tanhf(acc[3][r] + bc);

        const size_t sidx = ((size_t)dir * BB + b) * HH + h;
        const float C = f_ * Cst[sidx] + i_ * c_;
        Cst[sidx] = C;
        const float Hn = o_ * tanhf(C);
        Hwrite[sidx] = f2bf(Hn);

        const size_t midx = (size_t)b * (2 * HH) + (size_t)dir * HH + h;
        const float mv = maxbuf[midx];
        maxbuf[midx] = (Hn > mv) ? Hn : mv;
    }
}

// ---------------- 5) head: relu(pooled) @ W1 + b1 ----------------
__global__ __launch_bounds__(256) void head1_kernel(
    const float* __restrict__ maxbuf, const float* __restrict__ W1,
    const float* __restrict__ b1, float* __restrict__ out1)
{
    const int b = blockIdx.x;  // 0..127
    for (int j = threadIdx.x; j < HH / 2; j += blockDim.x) {
        float s = b1[j];
        for (int k = 0; k < 2 * HH; ++k) {
            float p = maxbuf[(size_t)b * (2 * HH) + k];
            p = p > 0.0f ? p : 0.0f;
            s += p * W1[(size_t)k * (HH / 2) + j];
        }
        out1[(size_t)b * (HH / 2) + j] = s;
    }
}

// ---------------- 6) head: out1 @ W2 + b2 ----------------
__global__ __launch_bounds__(256) void head2_kernel(
    const float* __restrict__ out1, const float* __restrict__ W2,
    const float* __restrict__ b2, float* __restrict__ out)
{
    const int tid = blockIdx.x * 256 + threadIdx.x;
    if (tid >= BB * LBL) return;
    const int b = tid / LBL, l = tid % LBL;
    float s = b2[l];
    for (int k = 0; k < HH / 2; ++k)
        s += out1[(size_t)b * (HH / 2) + k] * W2[(size_t)k * LBL + l];
    out[tid] = s;
}

// ---------------- launch ----------------
extern "C" void kernel_launch(void* const* d_in, const int* in_sizes, int n_in,
                              void* d_out, int out_size, void* d_ws, size_t ws_size,
                              hipStream_t stream) {
    const int*   inputs    = (const int*)  d_in[0];
    const float* emb_table = (const float*)d_in[1];
    const float* Wxf = (const float*)d_in[2];
    const float* Whf = (const float*)d_in[3];
    const float* bhf = (const float*)d_in[4];
    const float* Wxb = (const float*)d_in[5];
    const float* Whb = (const float*)d_in[6];
    const float* bhb = (const float*)d_in[7];
    const float* W1  = (const float*)d_in[8];
    const float* b1  = (const float*)d_in[9];
    const float* W2  = (const float*)d_in[10];
    const float* b2  = (const float*)d_in[11];
    float* out = (float*)d_out;

    // workspace layout (all 256B aligned)
    char* ws = (char*)d_ws;
    unsigned short* embB   = (unsigned short*)(ws);                        // 32 MB
    unsigned short* WT     = (unsigned short*)(ws + 33554432);             // 24 MB
    float*          Cst    = (float*)         (ws + 58720256);             // 1 MB
    unsigned short* Hb16   = (unsigned short*)(ws + 59768832);             // 1 MB (2 parity)
    float*          maxbuf = (float*)         (ws + 60817408);             // 1 MB
    float*          out1   = (float*)         (ws + 61865984);             // 256 KB

    gather_kernel<<<(TT * BB * EE) / 256, 256, 0, stream>>>(inputs, emb_table, embB);
    pack_kernel<<<(2 * G4H * KK) / 256, 256, 0, stream>>>(Wxf, Whf, Wxb, Whb, WT);
    init_kernel<<<(2 * 2 * BB * HH) / 256, 256, 0, stream>>>(Cst, Hb16, maxbuf);

    for (int t = 0; t < TT; ++t)
        step_kernel<<<128, 256, 0, stream>>>(embB, WT, bhf, bhb,
                                             Cst, Hb16, maxbuf, t);

    head1_kernel<<<BB, 256, 0, stream>>>(maxbuf, W1, b1, out1);
    head2_kernel<<<(BB * LBL + 255) / 256, 256, 0, stream>>>(out1, W2, b2, out);
}
